// TransformerEncoder_2860448219723
// MI455X (gfx1250) — matrix operand order repaired and checked by reading, not compiled, not run on hardware
//
#include <hip/hip_runtime.h>
#include <hip/hip_bf16.h>
#include <math.h>

#define DIM 1024
#define DEPTH 4
#define HEADS 16
#define DIM_HEAD 64
#define INNER 1024
#define MLP 4096
#define BATCH 8
#define SEQ 1024
#define ROWS (BATCH*SEQ)   // 8192

typedef __bf16 bf16_t;
typedef __attribute__((ext_vector_type(16))) __bf16 v16bf;
typedef __attribute__((ext_vector_type(8)))  float  v8f;
typedef __attribute__((ext_vector_type(4)))  unsigned int u32x4;
typedef __attribute__((ext_vector_type(2)))  unsigned int u32x2;
typedef __attribute__((ext_vector_type(8)))  int i32x8;
typedef __attribute__((ext_vector_type(4)))  int i32x4;

union FragBF { u32x4 q[2]; v16bf v; };
union Pack8  { u32x4 q; bf16_t e[8]; };
union Pack4  { u32x2 q; bf16_t e[4]; };

#if defined(__has_builtin)
#if __has_builtin(__builtin_amdgcn_tensor_load_to_lds) && __has_builtin(__builtin_amdgcn_s_wait_tensorcnt)
#define USE_TDM 1
#endif
#endif

// ---------------------------------------------------------------------------
// LayerNorm over DIM=1024, writes bf16. One row per block (256 thr = 8 waves).
// ---------------------------------------------------------------------------
__global__ __launch_bounds__(256) void layernorm_bf16_kernel(
    const float* __restrict__ x, const float* __restrict__ scale,
    const float* __restrict__ bias, bf16_t* __restrict__ out) {
  int row = blockIdx.x;
  int tid = threadIdx.x;
  const float* xr = x + (size_t)row * DIM;
  float4 v = *reinterpret_cast<const float4*>(xr + tid * 4);
  float s  = v.x + v.y + v.z + v.w;
  float s2 = v.x*v.x + v.y*v.y + v.z*v.z + v.w*v.w;
#pragma unroll
  for (int m = 16; m; m >>= 1) { s += __shfl_xor(s, m); s2 += __shfl_xor(s2, m); }
  __shared__ float rs[8], rs2[8], stats[2];
  int wave = tid >> 5, lane = tid & 31;
  if (lane == 0) { rs[wave] = s; rs2[wave] = s2; }
  __syncthreads();
  if (tid == 0) {
    float t = 0.f, t2 = 0.f;
    for (int i = 0; i < 8; i++) { t += rs[i]; t2 += rs2[i]; }
    float mu = t * (1.0f / DIM);
    float var = t2 * (1.0f / DIM) - mu * mu;
    stats[0] = mu; stats[1] = rsqrtf(var + 1e-5f);
  }
  __syncthreads();
  float mu = stats[0], inv = stats[1];
  int c = tid * 4;
  float xv[4] = { v.x, v.y, v.z, v.w };
  Pack4 p;
#pragma unroll
  for (int i = 0; i < 4; i++)
    p.e[i] = (bf16_t)((xv[i] - mu) * inv * scale[c + i] + bias[c + i]);
  *reinterpret_cast<u32x2*>(out + (size_t)row * DIM + c) = p.q;
}

// ---------------------------------------------------------------------------
// W f32 [K,N] -> Wt bf16 [N,K]  (so GEMM B-fragments are contiguous in k)
// ---------------------------------------------------------------------------
__global__ __launch_bounds__(256) void transpose_convert_kernel(
    const float* __restrict__ W, bf16_t* __restrict__ Wt, int K, int N) {
  __shared__ float tile[32][33];
  int k0 = blockIdx.x * 32, n0 = blockIdx.y * 32;
  int tx = threadIdx.x, ty = threadIdx.y;
#pragma unroll
  for (int i = 0; i < 4; i++)
    tile[ty + 8 * i][tx] = W[(size_t)(k0 + ty + 8 * i) * N + (n0 + tx)];
  __syncthreads();
#pragma unroll
  for (int i = 0; i < 4; i++)
    Wt[(size_t)(n0 + ty + 8 * i) * K + (k0 + tx)] = (bf16_t)tile[tx][ty + 8 * i];
}

// ---------------------------------------------------------------------------
// TDM: DMA one 128-row x 32-col bf16 tile (row stride = stride_elems) into
// LDS with hardware padding of 16B per 64B row -> 40-halfword LDS row pitch.
// Descriptor per CDNA5 ISA. Fully uniform args.
// This toolchain exposes the 6-arg builtin (g0 v4u, g1 v8i, g2 v4i, g3 v4i,
// extra v8i, cpol i32); groups 2/3 are unused for a 2D tile -> zeros.
// ---------------------------------------------------------------------------
#ifdef USE_TDM
__device__ __forceinline__ void tdm_load_tile_128x32(
    unsigned lds_addr, const bf16_t* gptr, int stride_elems) {
  unsigned long long ga = (unsigned long long)(size_t)gptr;
  u32x4 g0 = {
    1u,                                            // count=1, user, no gather
    lds_addr,                                      // lds_addr (bytes)
    (unsigned)(ga & 0xFFFFFFFFu),                  // global_addr[31:0]
    (unsigned)((ga >> 32) & 0x1FFFFFFu) | (2u << 30) // global_addr[56:32], type=2
  };
  i32x8 g1 = {
    (int)((1u << 16) |                             // data_size = 2 bytes
          (1u << 20) |                             // pad_enable
          (3u << 22) |                             // pad_interval: 16 DWORDs (64B)
          (3u << 25)),                             // pad_amount: 4 DWORDs (16B)
    (int)(32u << 16),                              // tensor_dim0 = 32 (lo16)
    (int)(128u << 16),                             // dim0 hi=0 | tensor_dim1 = 128
    (int)(32u << 16),                              // dim1 hi=0 | tile_dim0 = 32
    (int)128,                                      // tile_dim1 = 128 | tile_dim2 = 0
    (int)(unsigned)stride_elems,                   // tensor_dim0_stride lo32
    0, 0
  };
  i32x4 z4 = { 0, 0, 0, 0 };
  i32x8 z8 = { 0, 0, 0, 0, 0, 0, 0, 0 };
  __builtin_amdgcn_tensor_load_to_lds(g0, g1, z4, z4, z8, 0);
}
#endif

// ---------------------------------------------------------------------------
// WMMA bf16 GEMM:  out[M,N] = A[M,K] * Bt[N,K]^T  (+bias)(+GELU)(+residual)
// Block 128x128, 8 waves of 64x32, K-step 32.
// TDM path: double-buffered LDS, wave0 drives the Tensor Data Mover.
// ---------------------------------------------------------------------------
#define LDSPAD 40
__global__ __launch_bounds__(256, 2) void wmma_gemm_kernel(
    const bf16_t* __restrict__ A, const bf16_t* __restrict__ Bt,
    const float* __restrict__ bias, const float* __restrict__ residual,
    float* __restrict__ outF, bf16_t* __restrict__ outB,
    int M, int N, int K, int gelu) {
  int M0 = blockIdx.y * 128, N0 = blockIdx.x * 128;
  int tid = threadIdx.x, lane = tid & 31, wave = tid >> 5;
  int wm = (wave & 1) * 64, wn = (wave >> 1) * 32;
  int lm = lane & 15, lh = lane >> 4;
  v8f acc[4][2] = {};

#ifdef USE_TDM
  __shared__ __align__(16) bf16_t As[2][128 * LDSPAD];
  __shared__ __align__(16) bf16_t Bs[2][128 * LDSPAD];
  unsigned aLds0 = (unsigned)(size_t)&As[0][0];
  unsigned aLds1 = (unsigned)(size_t)&As[1][0];
  unsigned bLds0 = (unsigned)(size_t)&Bs[0][0];
  unsigned bLds1 = (unsigned)(size_t)&Bs[1][0];
  const bf16_t* aBase = A + (size_t)M0 * K;
  const bf16_t* bBase = Bt + (size_t)N0 * K;
  int nk = K >> 5;
  if (wave == 0) {
    tdm_load_tile_128x32(aLds0, aBase, K);
    tdm_load_tile_128x32(bLds0, bBase, K);
  }
  for (int ik = 0; ik < nk; ik++) {
    int buf = ik & 1;
    __syncthreads();  // all waves done reading buf^1 before TDM overwrites it
    if (wave == 0) {
      if (ik + 1 < nk) {
        tdm_load_tile_128x32(buf ? aLds0 : aLds1, aBase + (ik + 1) * 32, K);
        tdm_load_tile_128x32(buf ? bLds0 : bLds1, bBase + (ik + 1) * 32, K);
        __builtin_amdgcn_s_wait_tensorcnt(2);  // current buffer's 2 DMAs done
      } else {
        __builtin_amdgcn_s_wait_tensorcnt(0);
      }
    }
    __syncthreads();  // current buffer visible to all waves
    FragBF af[4];
#pragma unroll
    for (int i = 0; i < 4; i++) {
      const bf16_t* p = &As[buf][(wm + i * 16 + lm) * LDSPAD];
      af[i].q[0] = *reinterpret_cast<const u32x4*>(p + lh * 8);
      af[i].q[1] = *reinterpret_cast<const u32x4*>(p + 16 + lh * 8);
    }
    FragBF bfr[2];
#pragma unroll
    for (int j = 0; j < 2; j++) {
      const bf16_t* p = &Bs[buf][(wn + j * 16 + lm) * LDSPAD];
      bfr[j].q[0] = *reinterpret_cast<const u32x4*>(p + lh * 8);
      bfr[j].q[1] = *reinterpret_cast<const u32x4*>(p + 16 + lh * 8);
    }
#pragma unroll
    for (int i = 0; i < 4; i++)
#pragma unroll
      for (int j = 0; j < 2; j++)
        acc[i][j] = __builtin_amdgcn_wmma_f32_16x16x32_bf16(
            false, af[i].v, false, bfr[j].v, (short)0, acc[i][j], false, false);
  }
#else
  __shared__ __align__(16) bf16_t As[128 * LDSPAD];
  __shared__ __align__(16) bf16_t Bs[128 * LDSPAD];
  for (int kk = 0; kk < K; kk += 32) {
#pragma unroll
    for (int i = 0; i < 2; i++) {
      int idx = tid + 256 * i;
      int r = idx >> 2, cg = idx & 3;
      *reinterpret_cast<u32x4*>(&As[r * LDSPAD + cg * 8]) =
          *reinterpret_cast<const u32x4*>(A + (size_t)(M0 + r) * K + kk + cg * 8);
      *reinterpret_cast<u32x4*>(&Bs[r * LDSPAD + cg * 8]) =
          *reinterpret_cast<const u32x4*>(Bt + (size_t)(N0 + r) * K + kk + cg * 8);
    }
    __syncthreads();
    FragBF af[4];
#pragma unroll
    for (int i = 0; i < 4; i++) {
      const bf16_t* p = &As[(wm + i * 16 + lm) * LDSPAD];
      af[i].q[0] = *reinterpret_cast<const u32x4*>(p + lh * 8);
      af[i].q[1] = *reinterpret_cast<const u32x4*>(p + 16 + lh * 8);
    }
    FragBF bfr[2];
#pragma unroll
    for (int j = 0; j < 2; j++) {
      const bf16_t* p = &Bs[(wn + j * 16 + lm) * LDSPAD];
      bfr[j].q[0] = *reinterpret_cast<const u32x4*>(p + lh * 8);
      bfr[j].q[1] = *reinterpret_cast<const u32x4*>(p + 16 + lh * 8);
    }
#pragma unroll
    for (int i = 0; i < 4; i++)
#pragma unroll
      for (int j = 0; j < 2; j++)
        acc[i][j] = __builtin_amdgcn_wmma_f32_16x16x32_bf16(
            false, af[i].v, false, bfr[j].v, (short)0, acc[i][j], false, false);
    __syncthreads();
  }
#endif

  // Epilogue: C/D layout -> lane n = lm, row m = v + 8*lh
#pragma unroll
  for (int i = 0; i < 4; i++) {
#pragma unroll
    for (int j = 0; j < 2; j++) {
      int n = N0 + wn + j * 16 + lm;
      float bi = bias ? bias[n] : 0.0f;
#pragma unroll
      for (int v = 0; v < 8; v++) {
        int m = M0 + wm + i * 16 + v + 8 * lh;
        float val = acc[i][j][v] + bi;
        if (gelu) val = 0.5f * val * (1.0f + erff(val * 0.70710678118f));
        if (residual) val += residual[(size_t)m * N + n];
        if (outF) outF[(size_t)m * N + n] = val;
        if (outB) outB[(size_t)m * N + n] = (bf16_t)val;
      }
    }
  }
}

// ---------------------------------------------------------------------------
// Flash attention: block = (q-block 128 rows, head, batch); each wave owns a
// 16-row Q strip, streams 64-key tiles with online softmax, all matmuls WMMA.
// ---------------------------------------------------------------------------
__global__ __launch_bounds__(256, 2) void flash_attn_kernel(
    const bf16_t* __restrict__ Q, const bf16_t* __restrict__ Kb,
    const bf16_t* __restrict__ V, bf16_t* __restrict__ O) {
  __shared__ __align__(16) bf16_t Vt[64 * 72];      // V^T tile [d][j]
  __shared__ __align__(16) bf16_t Pl[8 * 16 * 72];  // per-wave P strips
  int h = blockIdx.y, bz = blockIdx.z;
  int tid = threadIdx.x, lane = tid & 31, wave = tid >> 5;
  int lm = lane & 15, lh = lane >> 4;
  int q0 = blockIdx.x * 128 + wave * 16;
  int ch = h * DIM_HEAD;
  size_t rowbase = (size_t)bz * SEQ;

  // Q strip fragments (16 x 64) stay in registers for the whole kernel
  FragBF qf[2];
  const bf16_t* qrow = Q + (rowbase + q0 + lm) * INNER + ch;
#pragma unroll
  for (int c = 0; c < 2; c++) {
    qf[c].q[0] = *reinterpret_cast<const u32x4*>(qrow + c * 32 + lh * 8);
    qf[c].q[1] = *reinterpret_cast<const u32x4*>(qrow + c * 32 + 16 + lh * 8);
  }
  float mrun[8], lrun[8];
#pragma unroll
  for (int v = 0; v < 8; v++) { mrun[v] = -INFINITY; lrun[v] = 0.f; }
  v8f o[4] = {};
  bf16_t* myP = &Pl[wave * 16 * 72];

  for (int j0 = 0; j0 < SEQ; j0 += 64) {
    // stage V tile transposed into LDS (coalesced global reads)
#pragma unroll
    for (int i = 0; i < 2; i++) {
      int idx = tid + 256 * i;
      int jr = idx >> 3, cg = idx & 7;
      Pack8 pk;
      pk.q = *reinterpret_cast<const u32x4*>(V + (rowbase + j0 + jr) * INNER + ch + cg * 8);
#pragma unroll
      for (int e = 0; e < 8; e++) Vt[(cg * 8 + e) * 72 + jr] = pk.e[e];
    }
    // prefetch next tile's K/V rows into the cache hierarchy
    if (j0 + 64 < SEQ) {
      int prow = tid & 63;
      __builtin_prefetch(Kb + (rowbase + j0 + 64 + prow) * INNER + ch, 0, 1);
      __builtin_prefetch(V + (rowbase + j0 + 64 + prow) * INNER + ch, 0, 1);
    }
    __syncthreads();

    // S = Q * K^T  (K rows [j][d] are already B^T-form: d contiguous)
    v8f st[4];
#pragma unroll
    for (int t = 0; t < 4; t++) {
      v8f s = {};
#pragma unroll
      for (int c = 0; c < 2; c++) {
        FragBF kf;
        const bf16_t* kr = Kb + (rowbase + j0 + t * 16 + lm) * INNER + ch + c * 32 + lh * 8;
        kf.q[0] = *reinterpret_cast<const u32x4*>(kr);
        kf.q[1] = *reinterpret_cast<const u32x4*>(kr + 16);
        s = __builtin_amdgcn_wmma_f32_16x16x32_bf16(
            false, qf[c].v, false, kf.v, (short)0, s, false, false);
      }
      st[t] = s * 0.125f;  // DIM_HEAD^-0.5
    }

    // online softmax: row max across 4 tiles + 16 lanes of this half
    float alpha[8], ps[8];
#pragma unroll
    for (int v = 0; v < 8; v++) {
      float m = fmaxf(fmaxf(st[0][v], st[1][v]), fmaxf(st[2][v], st[3][v]));
#pragma unroll
      for (int msk = 8; msk; msk >>= 1) m = fmaxf(m, __shfl_xor(m, msk));
      float mn = fmaxf(mrun[v], m);
      alpha[v] = __expf(mrun[v] - mn);
      mrun[v] = mn;
      ps[v] = 0.f;
    }
    // P = exp(S - m); spill to LDS to re-layout D-form -> A-form
#pragma unroll
    for (int t = 0; t < 4; t++) {
#pragma unroll
      for (int v = 0; v < 8; v++) {
        float p = __expf(st[t][v] - mrun[v]);
        ps[v] += p;
        myP[(v + 8 * lh) * 72 + t * 16 + lm] = (bf16_t)p;
      }
    }
#pragma unroll
    for (int v = 0; v < 8; v++) {
      float s = ps[v];
#pragma unroll
      for (int msk = 8; msk; msk >>= 1) s += __shfl_xor(s, msk);
      lrun[v] = lrun[v] * alpha[v] + s;
    }
#pragma unroll
    for (int t = 0; t < 4; t++)
#pragma unroll
      for (int v = 0; v < 8; v++) o[t][v] *= alpha[v];

    // O += P * V
#pragma unroll
    for (int c = 0; c < 2; c++) {
      FragBF pf;
      const bf16_t* pr = &myP[lm * 72 + c * 32];
      pf.q[0] = *reinterpret_cast<const u32x4*>(pr + lh * 8);
      pf.q[1] = *reinterpret_cast<const u32x4*>(pr + 16 + lh * 8);
#pragma unroll
      for (int t = 0; t < 4; t++) {
        FragBF vf;
        const bf16_t* vp = &Vt[(t * 16 + lm) * 72 + c * 32 + lh * 8];
        vf.q[0] = *reinterpret_cast<const u32x4*>(vp);
        vf.q[1] = *reinterpret_cast<const u32x4*>(vp + 16);
        o[t] = __builtin_amdgcn_wmma_f32_16x16x32_bf16(
            false, pf.v, false, vf.v, (short)0, o[t], false, false);
      }
    }
    __syncthreads();
  }
#pragma unroll
  for (int t = 0; t < 4; t++)
#pragma unroll
    for (int v = 0; v < 8; v++) {
      float val = o[t][v] / lrun[v];
      int m = q0 + v + 8 * lh;
      O[(rowbase + m) * INNER + ch + t * 16 + lm] = (bf16_t)val;
    }
}

// ---------------------------------------------------------------------------
extern "C" void kernel_launch(void* const* d_in, const int* in_sizes, int n_in,
                              void* d_out, int out_size, void* d_ws, size_t ws_size,
                              hipStream_t stream) {
  (void)in_sizes; (void)n_in; (void)out_size; (void)ws_size;
  const float* x_in   = (const float*)d_in[0];
  const float* ln_q_s = (const float*)d_in[1];
  const float* ln_q_b = (const float*)d_in[2];
  const float* ln_k_s = (const float*)d_in[3];
  const float* ln_k_b = (const float*)d_in[4];
  const float* ln_v_s = (const float*)d_in[5];
  const float* ln_v_b = (const float*)d_in[6];
  const float* Wq_all = (const float*)d_in[7];
  const float* Wk_all = (const float*)d_in[8];
  const float* Wv_all = (const float*)d_in[9];
  const float* Wo_all = (const float*)d_in[10];
  const float* bo_all = (const float*)d_in[11];
  const float* ln_f_s = (const float*)d_in[12];
  const float* ln_f_b = (const float*)d_in[13];
  const float* W1_all = (const float*)d_in[14];
  const float* b1_all = (const float*)d_in[15];
  const float* W2_all = (const float*)d_in[16];
  const float* b2_all = (const float*)d_in[17];

  char* ws = (char*)d_ws;
  size_t off = 0;
  auto carve = [&](size_t bytes) -> void* {
    void* p = ws + off;
    off += (bytes + 255) & ~(size_t)255;
    return p;
  };
  float*  xbuf = (float*)carve((size_t)ROWS * DIM * 4);
  bf16_t* xln  = (bf16_t*)carve((size_t)ROWS * DIM * 2);
  bf16_t* qb   = (bf16_t*)carve((size_t)ROWS * INNER * 2);
  bf16_t* kb   = (bf16_t*)carve((size_t)ROWS * INNER * 2);
  bf16_t* vb   = (bf16_t*)carve((size_t)ROWS * INNER * 2);
  bf16_t* ao   = (bf16_t*)carve((size_t)ROWS * INNER * 2);
  bf16_t* h1   = (bf16_t*)carve((size_t)ROWS * MLP * 2);
  bf16_t* WqT  = (bf16_t*)carve((size_t)DIM * INNER * 2);
  bf16_t* WkT  = (bf16_t*)carve((size_t)DIM * INNER * 2);
  bf16_t* WvT  = (bf16_t*)carve((size_t)DIM * INNER * 2);
  bf16_t* WoT  = (bf16_t*)carve((size_t)INNER * DIM * 2);
  bf16_t* W1T  = (bf16_t*)carve((size_t)DIM * MLP * 2);
  bf16_t* W2T  = (bf16_t*)carve((size_t)MLP * DIM * 2);

  (void)hipMemcpyAsync(xbuf, x_in, (size_t)ROWS * DIM * 4, hipMemcpyDeviceToDevice, stream);

  dim3 tb(32, 8);
  for (int l = 0; l < DEPTH; l++) {
    const float* Wq = Wq_all + (size_t)l * DIM * INNER;
    const float* Wk = Wk_all + (size_t)l * DIM * INNER;
    const float* Wv = Wv_all + (size_t)l * DIM * INNER;
    const float* Wo = Wo_all + (size_t)l * INNER * DIM;
    const float* W1 = W1_all + (size_t)l * DIM * MLP;
    const float* W2 = W2_all + (size_t)l * MLP * DIM;

    transpose_convert_kernel<<<dim3(DIM / 32, INNER / 32), tb, 0, stream>>>(Wq, WqT, DIM, INNER);
    transpose_convert_kernel<<<dim3(DIM / 32, INNER / 32), tb, 0, stream>>>(Wk, WkT, DIM, INNER);
    transpose_convert_kernel<<<dim3(DIM / 32, INNER / 32), tb, 0, stream>>>(Wv, WvT, DIM, INNER);
    transpose_convert_kernel<<<dim3(INNER / 32, DIM / 32), tb, 0, stream>>>(Wo, WoT, INNER, DIM);
    transpose_convert_kernel<<<dim3(DIM / 32, MLP / 32), tb, 0, stream>>>(W1, W1T, DIM, MLP);
    transpose_convert_kernel<<<dim3(MLP / 32, DIM / 32), tb, 0, stream>>>(W2, W2T, MLP, DIM);

    // Q/K/V projections (pre-norm, separate LNs), outputs bf16
    layernorm_bf16_kernel<<<ROWS, 256, 0, stream>>>(xbuf, ln_q_s + l * DIM, ln_q_b + l * DIM, xln);
    wmma_gemm_kernel<<<dim3(INNER / 128, ROWS / 128), 256, 0, stream>>>(
        xln, WqT, nullptr, nullptr, nullptr, qb, ROWS, INNER, DIM, 0);
    layernorm_bf16_kernel<<<ROWS, 256, 0, stream>>>(xbuf, ln_k_s + l * DIM, ln_k_b + l * DIM, xln);
    wmma_gemm_kernel<<<dim3(INNER / 128, ROWS / 128), 256, 0, stream>>>(
        xln, WkT, nullptr, nullptr, nullptr, kb, ROWS, INNER, DIM, 0);
    layernorm_bf16_kernel<<<ROWS, 256, 0, stream>>>(xbuf, ln_v_s + l * DIM, ln_v_b + l * DIM, xln);
    wmma_gemm_kernel<<<dim3(INNER / 128, ROWS / 128), 256, 0, stream>>>(
        xln, WvT, nullptr, nullptr, nullptr, vb, ROWS, INNER, DIM, 0);

    flash_attn_kernel<<<dim3(SEQ / 128, HEADS, BATCH), 256, 0, stream>>>(qb, kb, vb, ao);

    // output projection + bias + residual (in-place into xbuf)
    wmma_gemm_kernel<<<dim3(DIM / 128, ROWS / 128), 256, 0, stream>>>(
        ao, WoT, bo_all + (size_t)l * DIM, xbuf, xbuf, nullptr, ROWS, DIM, INNER, 0);

    // FFN: LN -> W1 + b1 -> exact GELU -> W2 + b2 + residual
    layernorm_bf16_kernel<<<ROWS, 256, 0, stream>>>(xbuf, ln_f_s + l * DIM, ln_f_b + l * DIM, xln);
    wmma_gemm_kernel<<<dim3(MLP / 128, ROWS / 128), 256, 0, stream>>>(
        xln, W1T, b1_all + (size_t)l * MLP, nullptr, nullptr, h1, ROWS, MLP, DIM, 1);
    wmma_gemm_kernel<<<dim3(DIM / 128, ROWS / 128), 256, 0, stream>>>(
        h1, W2T, b2_all + (size_t)l * DIM, xbuf, xbuf, nullptr, ROWS, DIM, MLP, 0);
  }

  (void)hipMemcpyAsync(d_out, xbuf, (size_t)ROWS * DIM * 4, hipMemcpyDeviceToDevice, stream);
}
